// local_pointnet_larger_ifnet_30159260352607
// MI455X (gfx1250) — matrix operand order repaired and checked by reading, not compiled, not run on hardware
//
#include <hip/hip_runtime.h>
#include <hip/hip_bf16.h>

typedef __attribute__((ext_vector_type(16))) _Float16 v16h;
typedef __attribute__((ext_vector_type(8)))  _Float16 v8h;
typedef __attribute__((ext_vector_type(8)))  float    v8f;

#define DEVI static __device__ __forceinline__

DEVI float lrelu(float x) { return x > 0.f ? x : 0.01f * x; }

DEVI v16h zero16h() {
  v16h z;
#pragma unroll
  for (int i = 0; i < 16; ++i) z[i] = (_Float16)0.f;
  return z;
}
DEVI v8h zero8h() {
  v8h z;
#pragma unroll
  for (int i = 0; i < 8; ++i) z[i] = (_Float16)0.f;
  return z;
}
DEVI v8f zero8f() {
  v8f z;
#pragma unroll
  for (int i = 0; i < 8; ++i) z[i] = 0.f;
  return z;
}
DEVI v16h cat8(v8h lo, v8h hi) {
  return __builtin_shufflevector(lo, hi, 0,1,2,3,4,5,6,7,8,9,10,11,12,13,14,15);
}

// ---- CDNA5 async copy: global -> LDS (ASYNCcnt), 16 bytes per lane ----------
DEVI void async_cp16(const _Float16* g, _Float16* l) {
  unsigned loff = (unsigned)(size_t)(const void*)l;  // low 32b of generic shared ptr = LDS offset
  asm volatile("global_load_async_to_lds_b128 %0, %1, off"
               :: "v"(loff), "v"(g) : "memory");
}
#if __has_builtin(__builtin_amdgcn_s_wait_asynccnt)
DEVI void wait_async() { __builtin_amdgcn_s_wait_asynccnt(0); }
#else
DEVI void wait_async() { asm volatile("s_wait_asynccnt 0x0" ::: "memory"); }
#endif

// A fragment (16x32 f16): lane l holds row l%16; K chunks [(l/16)*8 .. +7] and +16.
DEVI v16h loadA(const _Float16* row, int kbase, int lanehi) {
  int k0 = kbase + lanehi * 8;
  v8h lo = *(const v8h*)(row + k0);
  v8h hi = *(const v8h*)(row + k0 + 16);
  return cat8(lo, hi);
}
// B fragment (32x16 f16) from Wt[cout][K] row-major (stride Kp halves):
// lane l holds column N=l%16; K chunks [(l/16)*8 .. +7] and +16.
DEVI v16h loadB(const _Float16* wt, int Kp, int coutRow, int kbase, int lanehi) {
  const _Float16* p = wt + (long)coutRow * Kp + kbase + lanehi * 8;
  v8h lo = *(const v8h*)(p);
  v8h hi = *(const v8h*)(p + 16);
  return cat8(lo, hi);
}
DEVI v8f wmma16(v16h a, v16h b, v8f c) {
  return __builtin_amdgcn_wmma_f32_16x16x32_f16(false, a, false, b, (short)0, c, false, false);
}

// ---------------------------------------------------------------------------
// Fused double-GEMM: out = postact( lrelu(A @ W1 + b1) @ W2 + b2 [+resid] )
// MODE 0: A16[rows,128] plain (KP1=128)
// MODE 1: A row r = [xyz[r], 0...]                  (KP1=32)
// MODE 2: A row r = [feat16[idx[r]], xyz[r], 0...]  (KP1=160)
// POOL 1: max over groups of 8 rows then lrelu (out rows = rows/8)
// POOL 0: lrelu(h2 [+resid]) per row
// HASR/O16/O32: compile-time residual / f16-out / f32-out enables.
// Block = 256 threads (8 waves) = one strip of 16 A rows x 128 cols.
// A tile staged in LDS via one async B128 per thread (8x VMEM reuse).
// ---------------------------------------------------------------------------
template <int MODE, int POOL, bool HASR, bool O16, bool O32>
__global__ __launch_bounds__(256) void k_mlp2_t(
    const _Float16* __restrict__ A16, const int* __restrict__ idx,
    const float* __restrict__ xyz,
    const _Float16* __restrict__ W1t, const float* __restrict__ b1,
    const _Float16* __restrict__ W2t, const float* __restrict__ b2,
    const float* __restrict__ resid,
    _Float16* __restrict__ out16, float* __restrict__ out32)
{
  constexpr int KP1 = (MODE == 1) ? 32 : (MODE == 2 ? 160 : 128);
  constexpr int KSTEPS = KP1 / 32;
  __shared__ __align__(16) _Float16 al[16 * 128];
  __shared__ __align__(16) _Float16 hl[16 * 128];
  const int wid  = threadIdx.x >> 5;
  const int lane = threadIdx.x & 31;
  const int lo16 = lane & 15, hi = lane >> 4;
  const long strip = blockIdx.x;
  const long r0 = strip * 16;

  // ---- stage 16 A rows (feature part) into LDS: 256 threads x 1 async B128
  if constexpr (MODE != 1) {
    const int t = threadIdx.x;
    const int row = t >> 4, seg = t & 15;
    const long r = r0 + row;
    const _Float16* src;
    if constexpr (MODE == 2) src = A16 + (long)idx[r] * 128 + seg * 8;
    else                     src = A16 + r * 128 + seg * 8;
    async_cp16(src, &al[row * 128 + seg * 8]);
    wait_async();
  }
  __syncthreads();

  // ---- GEMM1: h = lrelu(A @ W1 + b1); wave computes cols [16*wid, 16*wid+16)
  v8f acc = zero8f();
  const long ar = r0 + lo16;                  // this lane's A row (A-layout)
  const _Float16* arow = &al[lo16 * 128];
#pragma unroll
  for (int ks = 0; ks < KSTEPS; ++ks) {
    const int kbase = ks * 32;
    v16h a;
    if constexpr (MODE == 1) {
      v16h z = zero16h();
      if (hi == 0) {
        z[0] = (_Float16)xyz[ar * 3 + 0];
        z[1] = (_Float16)xyz[ar * 3 + 1];
        z[2] = (_Float16)xyz[ar * 3 + 2];
      }
      a = z;
    } else if constexpr (MODE == 2) {
      if (ks == 4) {
        v16h z = zero16h();
        if (hi == 0) {
          z[0] = (_Float16)xyz[ar * 3 + 0];
          z[1] = (_Float16)xyz[ar * 3 + 1];
          z[2] = (_Float16)xyz[ar * 3 + 2];
        }
        a = z;
      } else {
        a = loadA(arow, kbase, hi);
      }
    } else {
      a = loadA(arow, kbase, hi);
    }
    v16h b = loadB(W1t, KP1, wid * 16 + lo16, kbase, hi);
    acc = wmma16(a, b, acc);
  }
  // bias + lrelu -> LDS (C layout: lane holds rows v+8*hi, col = 16*wid+l%16)
  {
    const int col = wid * 16 + lo16;
    const float bv = b1[col];
#pragma unroll
    for (int v = 0; v < 8; ++v)
      hl[(v + 8 * hi) * 128 + col] = (_Float16)lrelu(acc[v] + bv);
  }
  __syncthreads();

  // ---- GEMM2 from LDS
  v8f acc2 = zero8f();
  const _Float16* hrow = &hl[lo16 * 128];
#pragma unroll
  for (int ks = 0; ks < 4; ++ks) {
    v16h a = loadA(hrow, ks * 32, hi);
    v16h b = loadB(W2t, 128, wid * 16 + lo16, ks * 32, hi);
    acc2 = wmma16(a, b, acc2);
  }
  const int col = wid * 16 + lo16;
  const float bv = b2[col];
  if constexpr (POOL) {
    // rows v+8*hi for v=0..7 are exactly one K=8 pooling group per lane
    float m = acc2[0];
#pragma unroll
    for (int v = 1; v < 8; ++v) m = fmaxf(m, acc2[v]);
    const float o = lrelu(m + bv);
    const long orow = strip * 2 + hi;
    if constexpr (O16) out16[orow * 128 + col] = (_Float16)o;
    if constexpr (O32) out32[orow * 128 + col] = o;
  } else {
#pragma unroll
    for (int v = 0; v < 8; ++v) {
      const long orow = r0 + v + 8 * hi;
      float o = acc2[v] + bv;
      if constexpr (HASR) o += resid[orow * 128 + col];
      o = lrelu(o);
      if constexpr (O16) out16[orow * 128 + col] = (_Float16)o;
      if constexpr (O32) out32[orow * 128 + col] = o;
    }
  }
}

// ---------------------------------------------------------------------------
// Single GEMM (K=128): out = lrelu( A @ W + b + resid ), f16 + f32 outputs.
// (1x1x1 conv of the resnet_block_rec3.)  A staged in LDS via async copy.
// ---------------------------------------------------------------------------
__global__ __launch_bounds__(256) void k_gemm(
    const _Float16* __restrict__ A16, const _Float16* __restrict__ Wt,
    const float* __restrict__ bias, const float* __restrict__ resid,
    _Float16* __restrict__ out16, float* __restrict__ out32)
{
  __shared__ __align__(16) _Float16 al[16 * 128];
  const int wid = threadIdx.x >> 5, lane = threadIdx.x & 31;
  const int lo16 = lane & 15, hi = lane >> 4;
  const long r0 = (long)blockIdx.x * 16;
  {
    const int t = threadIdx.x;
    const int row = t >> 4, seg = t & 15;
    async_cp16(A16 + (r0 + row) * 128 + seg * 8, &al[row * 128 + seg * 8]);
    wait_async();
  }
  __syncthreads();
  const _Float16* arow = &al[lo16 * 128];
  v8f acc = zero8f();
#pragma unroll
  for (int ks = 0; ks < 4; ++ks) {
    v16h a = loadA(arow, ks * 32, hi);
    v16h b = loadB(Wt, 128, wid * 16 + lo16, ks * 32, hi);
    acc = wmma16(a, b, acc);
  }
  const int col = wid * 16 + lo16;
  const float bv = bias[col];
#pragma unroll
  for (int v = 0; v < 8; ++v) {
    const long orow = r0 + v + 8 * hi;
    float o = lrelu(acc[v] + bv + resid[orow * 128 + col]);
    out16[orow * 128 + col] = (_Float16)o;
    out32[orow * 128 + col] = o;
  }
}

// ---------------------------------------------------------------------------
// 3x3x3 conv, 128->128, SAME, over 32^3 grid. Block: 16 voxels (w-run) x 128
// couts; wave w -> couts [16w,16w+16). The 9 (kd,kh) neighbor row-runs
// (18 rows each, shared across the 3 kw taps and all 8 waves) are staged in
// LDS (40.5 KB) with async B128 copies; OOB rows zero-filled. Then 27x4 = 108
// fully-unrolled WMMAs off ds_load_b128.
// ---------------------------------------------------------------------------
__global__ __launch_bounds__(256) void k_conv3(
    const _Float16* __restrict__ xg, const _Float16* __restrict__ wt,
    const float* __restrict__ bias, _Float16* __restrict__ yg)
{
  __shared__ __align__(16) _Float16 al[9 * 18 * 128];   // 41,472 B
  const int wid = threadIdx.x >> 5, lane = threadIdx.x & 31;
  const int lo16 = lane & 15, hi = lane >> 4;
  const int b = blockIdx.x;            // 0..2047
  const int d = b >> 6;
  const int h = (b >> 1) & 31;
  const int w0 = (b & 1) * 16;

  // ---- stage halo: 9 slots x 18 rows x 16 segs = 2592 B128 chunks
  for (int i = 0; i < 11; ++i) {
    const int c = threadIdx.x + i * 256;
    if (c < 2592) {
      const int slot = c / 288, rem = c % 288;
      const int j = rem >> 4, seg = rem & 15;
      const int nd = d + slot / 3 - 1;
      const int nh = h + slot % 3 - 1;
      const int nw = w0 - 1 + j;
      _Float16* dst = &al[(slot * 18 + j) * 128 + seg * 8];
      if ((unsigned)nd < 32u && (unsigned)nh < 32u && (unsigned)nw < 32u)
        async_cp16(xg + (long)((nd * 32 + nh) * 32 + nw) * 128 + seg * 8, dst);
      else
        *(v8h*)dst = zero8h();
    }
  }
  wait_async();
  __syncthreads();

  v8f acc = zero8f();
#pragma unroll
  for (int s = 0; s < 9; ++s) {
#pragma unroll
    for (int kw = 0; kw < 3; ++kw) {
      const _Float16* arow = &al[(s * 18 + kw + lo16) * 128];
      const _Float16* wo = wt + (long)(s * 3 + kw) * 128 * 128;
#pragma unroll
      for (int ks = 0; ks < 4; ++ks) {
        v16h a  = loadA(arow, ks * 32, hi);
        v16h bb = loadB(wo, 128, wid * 16 + lo16, ks * 32, hi);
        acc = wmma16(a, bb, acc);
      }
    }
  }
  const int col = wid * 16 + lo16;
  const float bv = bias[col];
  const long base = (long)(d * 32 + h) * 32 + w0;
#pragma unroll
  for (int v = 0; v < 8; ++v)
    yg[(base + v + 8 * hi) * 128 + col] = (_Float16)lrelu(acc[v] + bv);
}

// ------------------------- small helper kernels -----------------------------
__global__ void k_prep_wt(const float* __restrict__ W, _Float16* __restrict__ Wt,
                          int Kdim, int Kp) {
  int i = blockIdx.x * blockDim.x + threadIdx.x;      // over 128*Kp
  if (i >= 128 * Kp) return;
  int n = i / Kp, k = i % Kp;
  Wt[i] = (k < Kdim) ? (_Float16)W[k * 128 + n] : (_Float16)0.f;
}
__global__ void k_prep_c3(const float* __restrict__ W, _Float16* __restrict__ Wt) {
  // W: [128o][128i][27] (OIDHW) -> Wt: [27][128o][128i]
  int i = blockIdx.x * blockDim.x + threadIdx.x;      // 27*128*128
  if (i >= 27 * 128 * 128) return;
  int o  = i / (128 * 128);
  int co = (i / 128) % 128;
  int ci = i % 128;
  Wt[i] = (_Float16)W[(co * 128 + ci) * 27 + o];
}
__global__ void k_cvt16(const float* __restrict__ W, _Float16* __restrict__ Wt, int n) {
  int i = blockIdx.x * blockDim.x + threadIdx.x;
  if (i < n) Wt[i] = (_Float16)W[i];
}
__global__ void k_zero16(_Float16* p, long n) {
  long i = (long)blockIdx.x * blockDim.x + threadIdx.x;
  if (i < n) p[i] = (_Float16)0.f;
}
__global__ void k_zero32(float* p, long n) {
  long i = (long)blockIdx.x * blockDim.x + threadIdx.x;
  if (i < n) p[i] = 0.f;
}
__global__ void k_init_mins(int* mins) {
  if (threadIdx.x < 3) mins[threadIdx.x] = 0x7fffffff;
}
__global__ void k_minred(const int* __restrict__ vi, int* mins, int M) {
  int i = blockIdx.x * blockDim.x + threadIdx.x;
  if (i < M * 3) atomicMin(&mins[i % 3], vi[i]);
}
__global__ void k_scatter(const int* __restrict__ vxyz, const int* __restrict__ mins,
                          const _Float16* __restrict__ s16, const float* __restrict__ s32,
                          _Float16* __restrict__ g16, float* __restrict__ g32, int M) {
  int i = blockIdx.x * blockDim.x + threadIdx.x;      // M*128
  if (i >= M * 128) return;
  int m = i >> 7, c = i & 127;
  int x = vxyz[m * 3 + 0] - mins[0];
  int y = vxyz[m * 3 + 1] - mins[1];
  int z = vxyz[m * 3 + 2] - mins[2];
  long cell = ((long)x * 32 + y) * 32 + z;
  g16[cell * 128 + c] = s16[i];
  g32[cell * 128 + c] = s32[i];
}
__global__ void k_gatherback(const int* __restrict__ vxyz, const int* __restrict__ mins,
                             const _Float16* __restrict__ g16, _Float16* __restrict__ o16, int M) {
  int i = blockIdx.x * blockDim.x + threadIdx.x;      // M*128
  if (i >= M * 128) return;
  int m = i >> 7, c = i & 127;
  int x = vxyz[m * 3 + 0] - mins[0];
  int y = vxyz[m * 3 + 1] - mins[1];
  int z = vxyz[m * 3 + 2] - mins[2];
  long cell = ((long)x * 32 + y) * 32 + z;
  o16[i] = g16[cell * 128 + c];
}
__global__ void k_head(const float* __restrict__ x, const float* __restrict__ Wb,
                       const float* __restrict__ bb, float* __restrict__ out, int M) {
  int i = blockIdx.x * blockDim.x + threadIdx.x;      // M*3
  if (i >= M * 3) return;
  int m = i / 3, j = i % 3;
  float s = bb[j];
  for (int c = 0; c < 128; ++c) s += x[m * 128 + c] * Wb[c * 3 + j];
  out[i] = 1.f / (1.f + expf(-s));
}

// ---------------------------------------------------------------------------
extern "C" void kernel_launch(void* const* d_in, const int* in_sizes, int n_in,
                              void* d_out, int out_size, void* d_ws, size_t ws_size,
                              hipStream_t stream)
{
  (void)in_sizes; (void)n_in; (void)out_size; (void)ws_size;
  const int N = 32768, M = 8192;
  const long NK = (long)N * 8, MK = (long)M * 8;
  const long GC = 32768;                               // 32^3 cells

  const int*   pc_idx = (const int*)d_in[0];
  const float* pc_xyz = (const float*)d_in[1];
  const int*   vxyz   = (const int*)d_in[2];
  const int*   v_idx  = (const int*)d_in[3];
  const float* v_xyz  = (const float*)d_in[4];
  const float* pcf_W1 = (const float*)d_in[5];
  const float* pcf_b1 = (const float*)d_in[6];
  const float* pcf_W2 = (const float*)d_in[7];
  const float* pcf_b2 = (const float*)d_in[8];
  const float* pr_W1  = (const float*)d_in[9];
  const float* pr_b1  = (const float*)d_in[10];
  const float* pr_W2  = (const float*)d_in[11];
  const float* pr_b2  = (const float*)d_in[12];
  const float* pcc_W1 = (const float*)d_in[13];
  const float* pcc_b1 = (const float*)d_in[14];
  const float* pcc_W2 = (const float*)d_in[15];
  const float* pcc_b2 = (const float*)d_in[16];
  const float* c3w    = (const float*)d_in[17];
  const float* c3b    = (const float*)d_in[18];
  const float* c1w    = (const float*)d_in[19];
  const float* c1b    = (const float*)d_in[20];
  const float* l1w    = (const float*)d_in[21];
  const float* l1b    = (const float*)d_in[22];
  const float* l2w    = (const float*)d_in[23];
  const float* l2b    = (const float*)d_in[24];
  const float* lbw    = (const float*)d_in[25];
  const float* lbb    = (const float*)d_in[26];

  char* ws = (char*)d_ws;
  size_t off = 0;
  auto alloc = [&](size_t bytes) -> char* {
    char* p = ws + off;
    off = (off + bytes + 255) & ~(size_t)255;
    return p;
  };
  // activations
  _Float16* x16  = (_Float16*)alloc((size_t)N * 128 * 2);   // feature table
  _Float16* t16  = (_Float16*)alloc((size_t)N * 128 * 2);   // conv-stage f16 out
  float*    s32  = (float*)   alloc((size_t)N * 128 * 4);   // conv-stage f32 out / lin out
  _Float16* xv16 = (_Float16*)alloc((size_t)M * 128 * 2);
  float*    xv32 = (float*)   alloc((size_t)M * 128 * 4);
  _Float16* g16  = (_Float16*)alloc((size_t)GC * 128 * 2);
  float*    g32  = (float*)   alloc((size_t)GC * 128 * 4);
  _Float16* y16  = (_Float16*)alloc((size_t)GC * 128 * 2);
  _Float16* gm16 = (_Float16*)alloc((size_t)M * 128 * 2);
  int*      mins = (int*)     alloc(16);
  // weights (f16, [cout][K])
  _Float16* pcfW1t = (_Float16*)alloc(128 * 32 * 2);
  _Float16* pcfW2t = (_Float16*)alloc(128 * 128 * 2);
  _Float16* prW1t  = (_Float16*)alloc((size_t)7 * 128 * 128 * 2);
  _Float16* prW2t  = (_Float16*)alloc((size_t)7 * 128 * 128 * 2);
  _Float16* pccW1t = (_Float16*)alloc((size_t)6 * 128 * 160 * 2);
  _Float16* pccW2t = (_Float16*)alloc((size_t)6 * 128 * 128 * 2);
  _Float16* c3t    = (_Float16*)alloc((size_t)8 * 27 * 128 * 128 * 2);
  _Float16* c1t    = (_Float16*)alloc((size_t)8 * 128 * 128 * 2);
  _Float16* l1t    = (_Float16*)alloc(128 * 128 * 2);
  _Float16* l2t    = (_Float16*)alloc(128 * 128 * 2);

  const int T = 256;
  auto cdiv = [](long a, long b) { return (int)((a + b - 1) / b); };

  // ---- weight prep (f32 -> f16, [cout][K] layout, K padded) ----
  k_prep_wt<<<cdiv(128 * 32, T), T, 0, stream>>>(pcf_W1, pcfW1t, 3, 32);
  k_prep_wt<<<cdiv(128 * 128, T), T, 0, stream>>>(pcf_W2, pcfW2t, 128, 128);
  for (int i = 0; i < 7; ++i) {
    k_prep_wt<<<cdiv(128 * 128, T), T, 0, stream>>>(pr_W1 + (long)i * 128 * 128, prW1t + (long)i * 128 * 128, 128, 128);
    k_prep_wt<<<cdiv(128 * 128, T), T, 0, stream>>>(pr_W2 + (long)i * 128 * 128, prW2t + (long)i * 128 * 128, 128, 128);
  }
  for (int i = 0; i < 6; ++i) {
    k_prep_wt<<<cdiv(128 * 160, T), T, 0, stream>>>(pcc_W1 + (long)i * 131 * 128, pccW1t + (long)i * 128 * 160, 131, 160);
    k_prep_wt<<<cdiv(128 * 128, T), T, 0, stream>>>(pcc_W2 + (long)i * 128 * 128, pccW2t + (long)i * 128 * 128, 128, 128);
  }
  for (int b = 0; b < 8; ++b)
    k_prep_c3<<<cdiv(27 * 128 * 128, T), T, 0, stream>>>(c3w + (long)b * 128 * 128 * 27, c3t + (long)b * 27 * 128 * 128);
  k_cvt16<<<cdiv(8 * 128 * 128, T), T, 0, stream>>>(c1w, c1t, 8 * 128 * 128);  // OI already [cout][cin]
  k_prep_wt<<<cdiv(128 * 128, T), T, 0, stream>>>(l1w, l1t, 128, 128);
  k_prep_wt<<<cdiv(128 * 128, T), T, 0, stream>>>(l2w, l2t, 128, 128);

  // ---- init grid scratch ----
  k_zero16<<<cdiv(GC * 128, T), T, 0, stream>>>(g16, GC * 128);
  k_zero32<<<cdiv(GC * 128, T), T, 0, stream>>>(g32, GC * 128);

  // ---- stage 1: pc_conv_first (3 -> 128 -> 128, maxpool K=8, lrelu) ----
  k_mlp2_t<1, 1, false, true, true><<<(int)(NK / 16), 256, 0, stream>>>(
      nullptr, nullptr, pc_xyz, pcfW1t, pcf_b1, pcfW2t, pcf_b2, nullptr, t16, s32);
  // pc_res[0]: reads t16/s32, writes feature table x16 (f16 only)
  k_mlp2_t<0, 0, true, true, false><<<N / 16, 256, 0, stream>>>(
      t16, nullptr, nullptr, prW1t, pr_b1, prW2t, pr_b2, s32, x16, nullptr);

  // ---- 6 x (gather-concat conv -> maxpool -> lrelu -> pc_res) ----
  for (int i = 0; i < 6; ++i) {
    const int*   gi = (i < 5) ? pc_idx : v_idx;
    const float* gx = (i < 5) ? pc_xyz : v_xyz;
    const long rows = (i < 5) ? NK : MK;
    // conv: gather x16 -> t16 (f16) + s32 (f32)
    k_mlp2_t<2, 1, false, true, true><<<(int)(rows / 16), 256, 0, stream>>>(
        x16, gi, gx,
        pccW1t + (long)i * 128 * 160, pcc_b1 + i * 128,
        pccW2t + (long)i * 128 * 128, pcc_b2 + i * 128, nullptr, t16, s32);
    const long orows = rows / 8;
    if (i < 5) {
      // pc_res: t16/s32 -> x16
      k_mlp2_t<0, 0, true, true, false><<<(int)(orows / 16), 256, 0, stream>>>(
          t16, nullptr, nullptr,
          prW1t + (long)(i + 1) * 128 * 128, pr_b1 + (i + 1) * 128,
          prW2t + (long)(i + 1) * 128 * 128, pr_b2 + (i + 1) * 128,
          s32, x16, nullptr);
    } else {
      // voxel pc_res: t16/s32 -> xv16 + xv32 (scatter needs both)
      k_mlp2_t<0, 0, true, true, true><<<(int)(orows / 16), 256, 0, stream>>>(
          t16, nullptr, nullptr,
          prW1t + (long)6 * 128 * 128, pr_b1 + 6 * 128,
          prW2t + (long)6 * 128 * 128, pr_b2 + 6 * 128,
          s32, xv16, xv32);
    }
  }

  // ---- scatter into 32^3 grid ----
  k_init_mins<<<1, 32, 0, stream>>>(mins);
  k_minred<<<cdiv((long)M * 3, T), T, 0, stream>>>(vxyz, mins, M);
  k_scatter<<<cdiv((long)M * 128, T), T, 0, stream>>>(vxyz, mins, xv16, xv32, g16, g32, M);

  // ---- 8 x resnet_block_rec3: y=lrelu(conv3x3x3(x)); x=lrelu(conv1(y)+x) ----
  for (int b = 0; b < 8; ++b) {
    k_conv3<<<2048, 256, 0, stream>>>(g16, c3t + (long)b * 27 * 128 * 128,
                                      c3b + b * 128, y16);
    k_gemm<<<(int)(GC / 16), 256, 0, stream>>>(y16, c1t + (long)b * 128 * 128,
                                               c1b + b * 128, g32, g16, g32);
  }

  // ---- gather back, lin1+lin2 fused, sigmoid head ----
  k_gatherback<<<cdiv((long)M * 128, T), T, 0, stream>>>(vxyz, mins, g16, gm16, M);
  k_mlp2_t<0, 0, false, false, true><<<M / 16, 256, 0, stream>>>(
      gm16, nullptr, nullptr, l1t, l1b, l2t, l2b, nullptr, nullptr, s32);
  k_head<<<cdiv((long)M * 3, T), T, 0, stream>>>(s32, lbw, lbb, (float*)d_out, M);
}
